// MyConv_4063039062211
// MI455X (gfx1250) — matrix operand order repaired and checked by reading, not compiled, not run on hardware
//
#include <hip/hip_runtime.h>

// ---- types ----------------------------------------------------------------
typedef __attribute__((ext_vector_type(16))) __bf16 v16bf;
typedef __attribute__((ext_vector_type(8)))  __bf16 v8bf;
typedef __attribute__((ext_vector_type(4)))  __bf16 v4bf;
typedef __attribute__((ext_vector_type(8)))  float  v8f;

// ---- problem constants (from reference) -----------------------------------
#define CIN   128
#define HW2   12544         // 112*112
#define WIN   112
#define COUT  256
#define OHW   55            // output H = W = (112-3)/2 + 1
#define OSQ   3025          // 55*55
#define KDIM  1152          // CIN*3*3
#define NTOT  96800         // 32 * 55 * 55

// ---- tiling ---------------------------------------------------------------
#define BM    128           // Cout rows per block
#define BN    128           // spatial cols per block
#define BK    32            // K per WMMA step (bf16 16x16x32)
#define NKC   (KDIM / BK)   // 36 K-chunks
#define LDA   40            // padded LDS row stride (halfwords): 80B = 20 banks
#define LDB   40

// A operand, 16-bit 16x32 layout: elem j <-> K = (j>>3)*16 + (lane>>4)*8 + (j&7)
static __device__ __forceinline__ v16bf op_a(const __bf16* rowbase, int khalf) {
    v8bf lo = *(const v8bf*)(rowbase + khalf);
    v8bf hi = *(const v8bf*)(rowbase + 16 + khalf);
    return __builtin_shufflevector(lo, hi, 0,1,2,3,4,5,6,7,8,9,10,11,12,13,14,15);
}

// B operand helper (unused; B is read via op_a): elem j <-> K = (lane>>4)*16 + j (contiguous)
static __device__ __forceinline__ v16bf op_b(const __bf16* rowbase, int lanehi) {
    const __bf16* p = rowbase + lanehi * 16;
    v8bf lo = *(const v8bf*)(p);
    v8bf hi = *(const v8bf*)(p + 8);
    return __builtin_shufflevector(lo, hi, 0,1,2,3,4,5,6,7,8,9,10,11,12,13,14,15);
}

static __device__ __forceinline__ v8f wmma_bf16(v16bf a, v16bf b, v8f c) {
    return __builtin_amdgcn_wmma_f32_16x16x32_bf16(
        false, a, false, b, (short)0, c, false, false);
}

__global__ __launch_bounds__(256)
void conv2d_s2_implicit_gemm_wmma(const float* __restrict__ x,
                                  const float* __restrict__ w,
                                  float* __restrict__ out) {
    __shared__ int    koff_tbl[KDIM];            //  4608 B: im2col k -> offset
    __shared__ __bf16 lds_a[2][BM * LDA];        // 2 x 10240 B (double buffer)
    __shared__ __bf16 lds_b[2][BN * LDB];        // 2 x 10240 B

    const int tid    = threadIdx.x;
    const int lane   = tid & 31;
    const int wave   = tid >> 5;
    const int m_wave = wave & 3;                 // 4 waves along M (32 rows each)
    const int n_wave = wave >> 2;                // 2 waves along N (64 cols each)
    const int lanehi = lane >> 4;
    const int lanelo = lane & 15;

    const int m0 = blockIdx.y * BM;              // Cout block base
    const int n0 = blockIdx.x * BN;              // spatial block base

    // ---- build koff table once: koff[k] = cin*HW + kh*112 + kw ------------
    for (int idx = tid; idx < KDIM; idx += 256) {
        const int cin = idx / 9;
        const int r9  = idx - cin * 9;
        const int kh  = r9 / 3;
        const int kw  = r9 - kh * 3;
        koff_tbl[idx] = cin * HW2 + kh * WIN + kw;
    }

    // ---- A (weight) loader: thread -> (row, 16-float half-row) ------------
    const int a_row  = tid >> 1;                 // 0..127
    const int a_half = (tid & 1) * 16;           // 0 or 16
    const float* a_src = w + (m0 + a_row) * KDIM + a_half;

    // ---- B (im2col) loader: thread -> 4 k-slots x 4 columns ---------------
    const int b_kq = (tid & 7) * 4;              // 0,4,..,28
    const int b_c0 = (tid >> 3) * 4;             // 0,4,..,124
    const float* xcol[4];
#pragma unroll
    for (int c = 0; c < 4; ++c) {
        int n_g = n0 + b_c0 + c;
        if (n_g >= NTOT) n_g = NTOT - 1;         // clamp: loads valid, unused
        const int bb  = n_g / OSQ;
        const int rem = n_g - bb * OSQ;
        const int oh  = rem / OHW;
        const int ow  = rem - oh * OHW;
        xcol[c] = x + (size_t)bb * (CIN * HW2) + (oh * 2) * WIN + (ow * 2);
    }

    const int khalf = lanehi * 8;

    v8f acc[2][4];
#pragma unroll
    for (int mt = 0; mt < 2; ++mt)
#pragma unroll
        for (int nt = 0; nt < 4; ++nt)
            acc[mt][nt] = (v8f){};

    float4 wreg[4];
    float  breg[4][4];                           // [col][kslot]

    // global -> regs for K-chunk kc (table-driven gathers, no divides)
    auto load_global = [&](int kc) {
        const float4* ap = (const float4*)(a_src + kc * BK);
        wreg[0] = ap[0]; wreg[1] = ap[1]; wreg[2] = ap[2]; wreg[3] = ap[3];
        const int4 ko = *(const int4*)&koff_tbl[kc * BK + b_kq];
#pragma unroll
        for (int c = 0; c < 4; ++c) {
            breg[c][0] = xcol[c][ko.x];
            breg[c][1] = xcol[c][ko.y];
            breg[c][2] = xcol[c][ko.z];
            breg[c][3] = xcol[c][ko.w];
        }
    };

    // regs -> LDS buffer p (hardware f32->bf16 packed converts)
    auto store_lds = [&](int p) {
        v8bf pa0 = { (__bf16)wreg[0].x, (__bf16)wreg[0].y, (__bf16)wreg[0].z, (__bf16)wreg[0].w,
                     (__bf16)wreg[1].x, (__bf16)wreg[1].y, (__bf16)wreg[1].z, (__bf16)wreg[1].w };
        v8bf pa1 = { (__bf16)wreg[2].x, (__bf16)wreg[2].y, (__bf16)wreg[2].z, (__bf16)wreg[2].w,
                     (__bf16)wreg[3].x, (__bf16)wreg[3].y, (__bf16)wreg[3].z, (__bf16)wreg[3].w };
        *(v8bf*)&lds_a[p][a_row * LDA + a_half]     = pa0;
        *(v8bf*)&lds_a[p][a_row * LDA + a_half + 8] = pa1;
#pragma unroll
        for (int c = 0; c < 4; ++c) {
            v4bf pb = { (__bf16)breg[c][0], (__bf16)breg[c][1],
                        (__bf16)breg[c][2], (__bf16)breg[c][3] };
            *(v4bf*)&lds_b[p][(b_c0 + c) * LDB + b_kq] = pb;
        }
    };

    // 8 WMMAs per wave from buffer p (EXEC all-ones: uniform control flow)
    auto mma = [&](int p) {
        const __bf16* ar0 = &lds_a[p][(m_wave * 32 + lanelo) * LDA];
        const __bf16* ar1 = ar0 + 16 * LDA;
        const __bf16* brb = &lds_b[p][(n_wave * 64 + lanelo) * LDB];
        v16bf a0 = op_a(ar0, khalf);
        v16bf a1 = op_a(ar1, khalf);
#pragma unroll
        for (int nt = 0; nt < 4; ++nt) {
            v16bf b = op_a(brb + nt * 16 * LDB, khalf);
            acc[0][nt] = wmma_bf16(a0, b, acc[0][nt]);
            acc[1][nt] = wmma_bf16(a1, b, acc[1][nt]);
        }
    };

    __syncthreads();           // koff table visible
    load_global(0);
    store_lds(0);
    __syncthreads();           // buffer 0 ready

    for (int kc = 0; kc < NKC; ++kc) {
        const int cur = kc & 1;
        if (kc + 1 < NKC) {
            load_global(kc + 1);                       // overlaps with MMAs
            __builtin_prefetch(a_src + (kc + 2) * BK, 0, 0);  // global_prefetch_b8
        }
        mma(cur);
        if (kc + 1 < NKC) {
            store_lds(cur ^ 1);                        // fill other buffer
            __syncthreads();                           // single barrier per step
        }
    }

    // ---- epilogue: D layout (elem r: M = r + 8*(lane>>4), N = lane&15) ----
    const int mrow = m0 + m_wave * 32 + 8 * lanehi;
#pragma unroll
    for (int nt = 0; nt < 4; ++nt) {
        const int ncol = n0 + n_wave * 64 + nt * 16 + lanelo;
        if (ncol < NTOT) {
            const int ob   = ncol / OSQ;
            const int orem = ncol - ob * OSQ;
            float* base = out + (size_t)ob * (COUT * OSQ) + orem;
#pragma unroll
            for (int r = 0; r < 8; ++r) {
                base[(size_t)(mrow + r)      * OSQ] = acc[0][nt][r];
                base[(size_t)(mrow + 16 + r) * OSQ] = acc[1][nt][r];
            }
        }
    }
}

extern "C" void kernel_launch(void* const* d_in, const int* in_sizes, int n_in,
                              void* d_out, int out_size, void* d_ws, size_t ws_size,
                              hipStream_t stream) {
    const float* x = (const float*)d_in[0];   // (32,128,112,112) f32
    const float* w = (const float*)d_in[1];   // (256,128,3,3)    f32
    float* out     = (float*)d_out;           // (32,256,55,55)   f32

    (void)in_sizes; (void)n_in; (void)out_size; (void)d_ws; (void)ws_size;

    dim3 grid((NTOT + BN - 1) / BN, COUT / BM);   // (757, 2)
    dim3 block(256);
    conv2d_s2_implicit_gemm_wmma<<<grid, block, 0, stream>>>(x, w, out);
}
